// SelfAttention_86311662780842
// MI455X (gfx1250) — matrix operand order repaired and checked by reading, not compiled, not run on hardware
//
#include <hip/hip_runtime.h>
#include <hip/hip_bf16.h>

typedef __bf16 bf16;
typedef __bf16 v16bf __attribute__((ext_vector_type(16)));
typedef __bf16 v8bf  __attribute__((ext_vector_type(8)));
typedef float  v8f   __attribute__((ext_vector_type(8)));
typedef float  v4f   __attribute__((ext_vector_type(4)));

#define B_  8
#define C_  512
#define CR_ 64
#define L_  2048
#define JSTEP 64

__device__ __forceinline__ v16bf join8(v8bf lo, v8bf hi) {
    v16bf r;
#pragma unroll
    for (int e = 0; e < 8; ++e) { r[e] = lo[e]; r[8 + e] = hi[e]; }
    return r;
}

__device__ __forceinline__ v8bf cvt8(v4f a, v4f b) {
    v8bf r;
#pragma unroll
    for (int e = 0; e < 4; ++e) { r[e] = (bf16)a[e]; r[4 + e] = (bf16)b[e]; }
    return r;
}

__device__ __forceinline__ v8f wmma_bf16(v16bf a, v16bf b, v8f c) {
    // D = A(16x32 bf16) x B(32x16 bf16) + C(16x16 f32)
    return __builtin_amdgcn_wmma_f32_16x16x32_bf16(false, a, false, b,
                                                   (short)0, c, false, false);
}

// ---------------------------------------------------------------------------
// Kernel 1: fused Q/K/V projection.  Out[o][i] = sum_c W[o][c] * x[b][c][i]
// o-blocks: 0 -> Wq rows, 1 -> Wk rows, 2..9 -> Wv rows.
// Stores Q^T, K^T as bf16 [B][L][64]; V as bf16 [B][C][L].
// ---------------------------------------------------------------------------
__global__ __launch_bounds__(256) void proj_kernel(
    const float* __restrict__ x,
    const float* __restrict__ wq, const float* __restrict__ bq,
    const float* __restrict__ wk, const float* __restrict__ bk,
    const float* __restrict__ wv, const float* __restrict__ bv,
    bf16* __restrict__ Qb, bf16* __restrict__ Kb, bf16* __restrict__ Vb)
{
    const int i0 = blockIdx.x * 64;
    const int ob = blockIdx.y;          // 0..9
    const int b  = blockIdx.z;
    const int t    = threadIdx.x;
    const int lane = t & 31;
    const int wid  = t >> 5;            // 0..7
    const int ot   = wid & 3;           // o-tile (16 rows)
    const int ih   = wid >> 2;          // i-half (2 i-tiles of 16)
    const int half = lane >> 4;
    const int l16  = lane & 15;

    __shared__ bf16 ldsXT[64 * 40];     // X^T tile [64 i][32 k], stride 40 (pad)

    const float* W; const float* bias;
    if (ob == 0)      { W = wq;                  bias = bq; }
    else if (ob == 1) { W = wk;                  bias = bk; }
    else              { W = wv + (ob - 2) * 64 * C_; bias = bv + (ob - 2) * 64; }

    v8f acc[2] = {};                    // two 16x16 C tiles (i-tiles ih*2, ih*2+1)

    const int aRow = ot * 16 + l16;     // A-operand row (o) for this lane
    const float* wrow = W + (size_t)aRow * C_;

    for (int k0 = 0; k0 < C_; k0 += 32) {
        // stage x[b][k0..k0+31][i0..i0+63] transposed into LDS as bf16
        {
            const int kk = t >> 3;          // 0..31
            const int ib = (t & 7) * 8;     // 0..56
            const float* xp = x + ((size_t)b * C_ + (k0 + kk)) * L_ + i0 + ib;
            v4f x0 = *(const v4f*)(xp);
            v4f x1 = *(const v4f*)(xp + 4);
            if (k0 + 32 < C_) __builtin_prefetch(xp + 32 * L_, 0, 3);
#pragma unroll
            for (int e = 0; e < 4; ++e) ldsXT[(ib + e) * 40 + kk]     = (bf16)x0[e];
#pragma unroll
            for (int e = 0; e < 4; ++e) ldsXT[(ib + 4 + e) * 40 + kk] = (bf16)x1[e];
        }
        __syncthreads();

        // A operand (16o x 32k): lane<16 -> k {0..7,16..23}; lane>=16 -> {8..15,24..31}
        const int kA = k0 + (half ? 8 : 0);
        v4f a0 = *(const v4f*)(wrow + kA);
        v4f a1 = *(const v4f*)(wrow + kA + 4);
        v4f a2 = *(const v4f*)(wrow + kA + 16);
        v4f a3 = *(const v4f*)(wrow + kA + 20);
        v16bf A = join8(cvt8(a0, a1), cvt8(a2, a3));

        // B operands (32k x 16i): lane<16 -> N=lane, k {0..7,16..23}; lane>=16 -> {8..15,24..31}
#pragma unroll
        for (int s = 0; s < 2; ++s) {
            const int irow = (ih * 2 + s) * 16 + l16;
            const bf16* p = &ldsXT[irow * 40 + (half ? 8 : 0)];
            v16bf Bop = join8(*(const v8bf*)p, *(const v8bf*)(p + 16));
            acc[s] = wmma_bf16(A, Bop, acc[s]);
        }
        __syncthreads();
    }

    // epilogue: bias + store (C layout: VGPR r -> row ot*16 + r + 8*half)
    if (ob <= 1) {
        bf16* dst = (ob == 0 ? Qb : Kb) + (size_t)b * L_ * CR_;
#pragma unroll
        for (int s = 0; s < 2; ++s) {
            const int i = i0 + (ih * 2 + s) * 16 + l16;
#pragma unroll
            for (int r = 0; r < 8; ++r) {
                const int oloc = ot * 16 + r + half * 8;
                dst[(size_t)i * CR_ + oloc] = (bf16)(acc[s][r] + bias[oloc]);
            }
        }
    } else {
        bf16* dst = Vb + ((size_t)b * C_ + (ob - 2) * 64) * L_;
#pragma unroll
        for (int s = 0; s < 2; ++s) {
            const int i = i0 + (ih * 2 + s) * 16 + l16;
#pragma unroll
            for (int r = 0; r < 8; ++r) {
                const int oloc = ot * 16 + r + half * 8;
                dst[(size_t)oloc * L_ + i] = (bf16)(acc[s][r] + bias[oloc]);
            }
        }
    }
}

// ---------------------------------------------------------------------------
// Kernel 2: flash attention. One workgroup per (b, 32 query rows).
// Waves 0/1: S = Q^T K (16i x 64j each) + online softmax, publish P via LDS.
// All 8 waves: O(64ch x 32i) += V * P^T; epilogue gamma*O/l + x.
// ---------------------------------------------------------------------------
__global__ __launch_bounds__(256) void attn_kernel(
    const bf16* __restrict__ Qb, const bf16* __restrict__ Kb,
    const bf16* __restrict__ Vb,
    const float* __restrict__ x, const float* __restrict__ gamma,
    float* __restrict__ out)
{
    const int i0 = blockIdx.x * 32;
    const int b  = blockIdx.y;
    const int t    = threadIdx.x;
    const int lane = t & 31;
    const int w    = t >> 5;            // wave: channels [64w, 64w+64)
    const int half = lane >> 4;
    const int l16  = lane & 15;

    __shared__ bf16  ldsP[2][32][JSTEP]; // P tile, double buffered (8KB)
    __shared__ float ldsS[2][32];        // per-row rescale factors
    __shared__ float ldsL[32];           // final softmax denominators

    v8f acc[4][2] = {};                 // 4 c-tiles x 2 i-tiles

    const bool sWave = (w < 2);
    v16bf Aq0 = {}, Aq1 = {};
    float m_st[8], l_st[8];
    if (sWave) {
        // A operand: Q^T rows i0 + w*16 .. +15, K-dim = d (64 -> two operands)
        const int irow = i0 + w * 16 + l16;
        const bf16* qp = Qb + ((size_t)b * L_ + irow) * CR_;
        const int kA = half ? 16 : 0;
        Aq0 = join8(*(const v8bf*)(qp + kA),      *(const v8bf*)(qp + kA + 8));
        Aq1 = join8(*(const v8bf*)(qp + 32 + kA), *(const v8bf*)(qp + 32 + kA + 8));
#pragma unroll
        for (int r = 0; r < 8; ++r) { m_st[r] = -3.0e38f; l_st[r] = 0.0f; }
    }

    for (int j0 = 0; j0 < L_; j0 += JSTEP) {
        const int buf = (j0 >> 6) & 1;

        if (sWave) {
            v8f S[4] = {};
#pragma unroll
            for (int jt = 0; jt < 4; ++jt) {
                // B operand: K (64d x 16j), read K^T[j][d] contiguous in d
                const int jcol = j0 + jt * 16 + l16;
                const bf16* kp = Kb + ((size_t)b * L_ + jcol) * CR_;
                const int kB = half ? 16 : 0;
                v16bf Bk0 = join8(*(const v8bf*)(kp + kB),
                                  *(const v8bf*)(kp + kB + 8));
                v16bf Bk1 = join8(*(const v8bf*)(kp + 32 + kB),
                                  *(const v8bf*)(kp + 32 + kB + 8));
                S[jt] = wmma_bf16(Aq0, Bk0, S[jt]);
                S[jt] = wmma_bf16(Aq1, Bk1, S[jt]);
            }
            if (j0 + JSTEP < L_)
                __builtin_prefetch(Kb + ((size_t)b * L_ + (j0 + JSTEP + lane)) * CR_, 0, 3);

            // online softmax: row i = w*16 + r + 8*half; its 16 j-cols are the lanes
#pragma unroll
            for (int r = 0; r < 8; ++r) {
                float v0 = S[0][r], v1 = S[1][r], v2 = S[2][r], v3 = S[3][r];
                float mt = fmaxf(fmaxf(v0, v1), fmaxf(v2, v3));
#pragma unroll
                for (int m = 8; m >= 1; m >>= 1)
                    mt = fmaxf(mt, __shfl_xor(mt, m, 32));
                const float mnew = fmaxf(m_st[r], mt);
                const float sc = __expf(m_st[r] - mnew);
                const float p0 = __expf(v0 - mnew);
                const float p1 = __expf(v1 - mnew);
                const float p2 = __expf(v2 - mnew);
                const float p3 = __expf(v3 - mnew);
                float rs = (p0 + p1) + (p2 + p3);
#pragma unroll
                for (int m = 8; m >= 1; m >>= 1)
                    rs += __shfl_xor(rs, m, 32);
                l_st[r] = l_st[r] * sc + rs;
                m_st[r] = mnew;
                const int iloc = w * 16 + r + half * 8;
                ldsP[buf][iloc][l16]      = (bf16)p0;
                ldsP[buf][iloc][16 + l16] = (bf16)p1;
                ldsP[buf][iloc][32 + l16] = (bf16)p2;
                ldsP[buf][iloc][48 + l16] = (bf16)p3;
                if (l16 == 0) ldsS[buf][iloc] = sc;
            }
        }
        __syncthreads();

        // consume: rescale accumulators once, then O += V * P^T over 64 j
        float f[2];
#pragma unroll
        for (int s = 0; s < 2; ++s) f[s] = ldsS[buf][s * 16 + l16];
#pragma unroll
        for (int ct = 0; ct < 4; ++ct)
#pragma unroll
            for (int s = 0; s < 2; ++s)
#pragma unroll
                for (int r = 0; r < 8; ++r) acc[ct][s][r] *= f[s];

        v16bf Bp[2][2];
#pragma unroll
        for (int s = 0; s < 2; ++s)
#pragma unroll
            for (int jh = 0; jh < 2; ++jh) {
                // B operand: P^T (32j x 16i) == read P[i][j] contiguous in j
                const bf16* pp = &ldsP[buf][s * 16 + l16][jh * 32 + (half ? 8 : 0)];
                Bp[s][jh] = join8(*(const v8bf*)pp, *(const v8bf*)(pp + 16));
            }

#pragma unroll
        for (int ct = 0; ct < 4; ++ct) {
            // A operands: V (16c x 32j) row-major in j, two j-halves
            const int c = w * 64 + ct * 16 + l16;
            const bf16* vp = Vb + ((size_t)b * C_ + c) * L_ + j0;
            const int jo = half ? 8 : 0;
            v16bf Av0 = join8(*(const v8bf*)(vp + jo),
                              *(const v8bf*)(vp + jo + 16));
            v16bf Av1 = join8(*(const v8bf*)(vp + 32 + jo),
                              *(const v8bf*)(vp + 32 + jo + 16));
            if (j0 + JSTEP < L_) __builtin_prefetch(vp + JSTEP, 0, 3);
#pragma unroll
            for (int s = 0; s < 2; ++s) {
                acc[ct][s] = wmma_bf16(Av0, Bp[s][0], acc[ct][s]);
                acc[ct][s] = wmma_bf16(Av1, Bp[s][1], acc[ct][s]);
            }
        }
    }

    // publish softmax denominators
    if (sWave) {
#pragma unroll
        for (int r = 0; r < 8; ++r)
            if (l16 == 0) ldsL[w * 16 + r + half * 8] = l_st[r];
    }
    __syncthreads();

    const float g = gamma[0];
    float linv[2];
#pragma unroll
    for (int s = 0; s < 2; ++s) linv[s] = 1.0f / ldsL[s * 16 + l16];

#pragma unroll
    for (int ct = 0; ct < 4; ++ct) {
#pragma unroll
        for (int s = 0; s < 2; ++s) {
#pragma unroll
            for (int r = 0; r < 8; ++r) {
                const int c = w * 64 + ct * 16 + r + half * 8;
                const int i = i0 + s * 16 + l16;
                const size_t idx = ((size_t)b * C_ + c) * L_ + i;
                out[idx] = g * acc[ct][s][r] * linv[s] + x[idx];
            }
        }
    }
}

extern "C" void kernel_launch(void* const* d_in, const int* in_sizes, int n_in,
                              void* d_out, int out_size, void* d_ws, size_t ws_size,
                              hipStream_t stream) {
    (void)in_sizes; (void)n_in; (void)out_size; (void)ws_size;
    const float* x   = (const float*)d_in[0];
    const float* wq  = (const float*)d_in[1];
    const float* bq  = (const float*)d_in[2];
    const float* wk  = (const float*)d_in[3];
    const float* bk  = (const float*)d_in[4];
    const float* wvp = (const float*)d_in[5];
    const float* bvp = (const float*)d_in[6];
    const float* gam = (const float*)d_in[7];

    // workspace: Qb (2MB) | Kb (2MB) | Vb (16MB), all bf16
    bf16* Qb = (bf16*)d_ws;
    bf16* Kb = Qb + (size_t)B_ * L_ * CR_;
    bf16* Vb = Kb + (size_t)B_ * L_ * CR_;

    dim3 g1(L_ / 64, (2 * CR_ + C_) / 64, B_);   // (32, 10, 8)
    proj_kernel<<<g1, 256, 0, stream>>>(x, wq, bq, wk, bk, wvp, bvp, Qb, Kb, Vb);

    dim3 g2(L_ / 32, B_);                        // (64, 8)
    attn_kernel<<<g2, 256, 0, stream>>>(Qb, Kb, Vb, x, gam, (float*)d_out);
}
